// EVTSR_22771916603484
// MI455X (gfx1250) — matrix operand order repaired and checked by reading, not compiled, not run on hardware
//
#include <hip/hip_runtime.h>
#include <hip/hip_bf16.h>

// ---------------------------------------------------------------------------
// CDNA5 / gfx1250. Constants: B=1, Cin=64, H=W=128 (P=16384), Ch=128, K=5.
// Heavy GEMMs: v_wmma_f32_16x16x32_bf16 (bf16 A/B, fp32 accumulate).
// Activation (B-matrix) tiles staged in LDS, transposed + padded so every
// fragment is two ds_load_b128 per lane (bank-conflict free).
// ---------------------------------------------------------------------------

typedef __bf16 bf16_t;
typedef __attribute__((ext_vector_type(16))) __bf16 v16bf;
typedef __attribute__((ext_vector_type(8)))  __bf16 v8bf;
typedef __attribute__((ext_vector_type(8)))  float  v8f;
typedef __attribute__((ext_vector_type(4)))  float  v4f;

#define PIX   16384   // H*W
#define HH    128
#define WW    128
#define CH    128
#define CIN   64

__device__ __forceinline__ v8f wmma_bf16(v16bf a, v16bf b, v8f c) {
  return __builtin_amdgcn_wmma_f32_16x16x32_bf16(
      /*neg_a=*/false, a, /*neg_b=*/false, b,
      /*c_mod=*/(short)0, c, /*reuse_a=*/false, /*reuse_b=*/false);
}

union FragU { v16bf v; v8bf h[2]; };

// A fragment (16x32 bf16). rowp -> (row m, k0) of row-major weight matrix,
// 16B aligned. Lane (m = lane&15, h = lane>>4) holds K = {h*16..h*16+15}:
// two contiguous 16-byte global loads.
__device__ __forceinline__ v16bf load_a_ptr(const bf16_t* rowp, int h) {
  FragU u;
  u.h[0] = *(const v8bf*)(rowp + h * 16);
  u.h[1] = *(const v8bf*)(rowp + h * 16 + 8);
  return u.v;
}

// B fragment (32x16 bf16) from a transposed LDS tile s[n][KPAD].
// Lane (n, h): elements K = k0 + h*16 + e are contiguous -> two ds_load_b128.
__device__ __forceinline__ v16bf frag_b_lds(const bf16_t* row) {
  FragU u;
  u.h[0] = *(const v8bf*)(row);
  u.h[1] = *(const v8bf*)(row + 8);
  return u.v;
}

// ---------------------------------------------------------------------------
// Conversion / repack kernels
// ---------------------------------------------------------------------------
__global__ void k_f2bf(const float* __restrict__ s, bf16_t* __restrict__ d, int n) {
  for (int i = blockIdx.x * blockDim.x + threadIdx.x; i < n;
       i += gridDim.x * blockDim.x)
    d[i] = (bf16_t)s[i];
}

// df1_w [Co=128][Ci=128][3][3] fp32  ->  w9[t=kh*3+kw][Co][Ci] bf16
__global__ void k_repack_df1(const float* __restrict__ w, bf16_t* __restrict__ w9) {
  int i = blockIdx.x * blockDim.x + threadIdx.x;  // over 128*128*9
  if (i >= CH * CH * 9) return;
  int t  = i % 9;
  int ci = (i / 9) % CH;
  int co = i / (9 * CH);
  w9[((size_t)t * CH + co) * CH + ci] = (bf16_t)w[i];
}

// ---------------------------------------------------------------------------
// feat GEMM: F[128,P] = conv_w[128,64] x X[64,P]   (K=64 -> 2 WMMAs/tile)
// 256 thr = 8 waves; wave = channel tile, blockIdx = pixel tile.
// ---------------------------------------------------------------------------
__global__ void __launch_bounds__(256)
k_gemm_feat(const bf16_t* __restrict__ Wc, const bf16_t* __restrict__ Xb,
            float* __restrict__ F) {
  const int KPAD = 72;                       // 144B rows: aligned, conflict-free
  __shared__ __align__(16) bf16_t sX[16 * KPAD];
  int p0 = blockIdx.x * 16;
  for (int i = threadIdx.x; i < CIN * 16; i += 256) {  // i = k*16 + n
    int k = i >> 4, n = i & 15;
    sX[n * KPAD + k] = Xb[(size_t)k * PIX + p0 + n];
  }
  __syncthreads();

  int lane = threadIdx.x & 31;
  int mt   = threadIdx.x >> 5;
  int m = lane & 15, h = lane >> 4, n = m;
  v8f acc = {};
#pragma unroll
  for (int kk = 0; kk < 2; ++kk) {
    v16bf a = load_a_ptr(Wc + (size_t)(mt * 16 + m) * CIN + kk * 32, h);
    v16bf b = frag_b_lds(sX + n * KPAD + kk * 32 + h * 16);
    acc = wmma_bf16(a, b, acc);
  }
#pragma unroll
  for (int r = 0; r < 8; ++r)
    F[(size_t)(mt * 16 + 8 * h + r) * PIX + p0 + n] = acc[r];
}

// ---------------------------------------------------------------------------
// LayerNorm over channels (biased var, eps=1e-5). In-place fp32 + bf16 copy.
// ---------------------------------------------------------------------------
__global__ void k_ln(float* __restrict__ F, const float* __restrict__ g,
                     const float* __restrict__ bta, bf16_t* __restrict__ Ybf) {
  int p = blockIdx.x * blockDim.x + threadIdx.x;
  if (p >= PIX) return;
  float s = 0.f, s2 = 0.f;
  for (int c = 0; c < CH; ++c) {
    float v = F[(size_t)c * PIX + p];
    s += v; s2 += v * v;
  }
  float mean = s * (1.0f / CH);
  float var  = s2 * (1.0f / CH) - mean * mean;
  float rinv = rsqrtf(var + 1e-5f);
  for (int c = 0; c < CH; ++c) {
    float v = (F[(size_t)c * PIX + p] - mean) * rinv * g[c] + bta[c];
    F[(size_t)c * PIX + p]   = v;
    Ybf[(size_t)c * PIX + p] = (bf16_t)v;
  }
}

// ---------------------------------------------------------------------------
// conv3x3 + ReLU as 9 shifted K=128 GEMMs (36 WMMAs / 16x16 tile).
// Stages the full 3x18 pixel halo (zero padded) transposed in LDS:
//   sX[(ry*18+rx) * 136 + ci]
// ---------------------------------------------------------------------------
__global__ void __launch_bounds__(256)
k_conv3x3(const bf16_t* __restrict__ W9, const bf16_t* __restrict__ Xbf,
          bf16_t* __restrict__ G) {
  const int KPAD = 136;                      // 272B rows
  __shared__ __align__(16) bf16_t sX[54 * KPAD];   // 14.3 KB
  int p0  = blockIdx.x * 16;
  int py  = p0 >> 7;
  int px0 = p0 & (WW - 1);

  for (int i = threadIdx.x; i < CH * 54; i += 256) {  // i = k*54 + j
    int k = i / 54, j = i % 54;
    int ry = j / 18, rx = j % 18;
    int y = py + ry - 1, x = px0 + rx - 1;
    bool ok = (y >= 0) && (y < HH) && (x >= 0) && (x < WW);
    sX[j * KPAD + k] = ok ? Xbf[(size_t)k * PIX + y * WW + x] : (bf16_t)0.0f;
  }
  __syncthreads();

  int lane = threadIdx.x & 31;
  int mt   = threadIdx.x >> 5;
  int m = lane & 15, h = lane >> 4, n = m;
  v8f acc = {};
#pragma unroll
  for (int t = 0; t < 9; ++t) {
    int dy = t / 3, dx = t % 3;              // 0..2 each
    int j = dy * 18 + (n + dx);              // halo-relative row
    const bf16_t* Wt = W9 + (size_t)t * CH * CH;
#pragma unroll
    for (int kk = 0; kk < 4; ++kk) {
      v16bf a = load_a_ptr(Wt + (size_t)(mt * 16 + m) * CH + kk * 32, h);
      v16bf b = frag_b_lds(sX + j * KPAD + kk * 32 + h * 16);
      acc = wmma_bf16(a, b, acc);
    }
  }
#pragma unroll
  for (int r = 0; r < 8; ++r) {
    float v = acc[r] > 0.f ? acc[r] : 0.f;
    G[(size_t)(mt * 16 + 8 * h + r) * PIX + p0 + n] = (bf16_t)v;
  }
}

// ---------------------------------------------------------------------------
// FUSED: filter = df2_w[3200,128] x G[128,P] (never materialized), then
// en[c,p] = sum_{k=0..24} feat[c, p+off(k)] * filt[c*25+k, p].
// Stages: G tile (transposed bf16, B frags reused across all 25 taps) and
// the 5x20x128 fp32 feat halo (zero padded) -> epilogue is ds_load_b128 pairs.
// 100 WMMAs per 16ch x 16px tile.
// ---------------------------------------------------------------------------
__global__ void __launch_bounds__(256)
k_dynfilter(const bf16_t* __restrict__ Wdf2, const bf16_t* __restrict__ G,
            const float* __restrict__ feat,
            float* __restrict__ outF, bf16_t* __restrict__ outB) {
  const int KPAD = 136;
  const int FPAD = 132;                       // 528B rows
  __shared__ __align__(16) bf16_t sG[16 * KPAD];   // 4.3 KB
  __shared__ __align__(16) float  sF[100 * FPAD]; // 52.8 KB
  int p0  = blockIdx.x * 16;
  int py  = p0 >> 7;
  int px0 = p0 & (WW - 1);

  for (int i = threadIdx.x; i < CH * 16; i += 256) {  // i = k*16 + n
    int k = i >> 4, n = i & 15;
    sG[n * KPAD + k] = G[(size_t)k * PIX + p0 + n];
  }
  for (int i = threadIdx.x; i < CH * 100; i += 256) { // i = c*100 + j
    int c = i / 100, j = i % 100;
    int ry = j / 20, rx = j % 20;
    int y = py + ry - 2, x = px0 + rx - 2;
    bool ok = (y >= 0) && (y < HH) && (x >= 0) && (x < WW);
    sF[j * FPAD + c] = ok ? feat[(size_t)c * PIX + y * WW + x] : 0.f;
  }
  __syncthreads();

  int lane = threadIdx.x & 31;
  int c0   = (threadIdx.x >> 5) * 16;
  int m = lane & 15, h = lane >> 4, n = m;

  v16bf b0 = frag_b_lds(sG + n * KPAD + 0  + h * 16);
  v16bf b1 = frag_b_lds(sG + n * KPAD + 32 + h * 16);
  v16bf b2 = frag_b_lds(sG + n * KPAD + 64 + h * 16);
  v16bf b3 = frag_b_lds(sG + n * KPAD + 96 + h * 16);

  float accO[8] = {0.f, 0.f, 0.f, 0.f, 0.f, 0.f, 0.f, 0.f};

  for (int k = 0; k < 25; ++k) {
    int dy = k / 5, dx = k % 5;               // 0..4 each
    const bf16_t* rowp = Wdf2 + ((size_t)(c0 + m) * 25 + k) * CH;
    v8f acc = {};
    acc = wmma_bf16(load_a_ptr(rowp + 0,  h), b0, acc);
    acc = wmma_bf16(load_a_ptr(rowp + 32, h), b1, acc);
    acc = wmma_bf16(load_a_ptr(rowp + 64, h), b2, acc);
    acc = wmma_bf16(load_a_ptr(rowp + 96, h), b3, acc);
    const float* fr = sF + (dy * 20 + n + dx) * FPAD + c0 + 8 * h;
    v4f f0 = *(const v4f*)(fr);
    v4f f1 = *(const v4f*)(fr + 4);
#pragma unroll
    for (int r = 0; r < 4; ++r) {
      accO[r]     += acc[r]     * f0[r];
      accO[r + 4] += acc[r + 4] * f1[r];
    }
  }
#pragma unroll
  for (int r = 0; r < 8; ++r) {
    size_t idx = (size_t)(c0 + 8 * h + r) * PIX + p0 + n;
    if (outF) outF[idx] = accO[r];
    if (outB) outB[idx] = (bf16_t)accO[r];
  }
}

// ---------------------------------------------------------------------------
// gate GEMM + sigmoid + residual fuse:
// fused = rgb_en * (1 + sigmoid(sp_w x ev_en)) + rgb_feat
// ---------------------------------------------------------------------------
__global__ void __launch_bounds__(256)
k_fuse(const bf16_t* __restrict__ Wsp, const bf16_t* __restrict__ EnEvB,
       const float* __restrict__ EnRgb, const float* __restrict__ FeatRgb,
       float* __restrict__ Fused) {
  const int KPAD = 136;
  __shared__ __align__(16) bf16_t sX[16 * KPAD];
  int p0 = blockIdx.x * 16;
  for (int i = threadIdx.x; i < CH * 16; i += 256) {
    int k = i >> 4, n = i & 15;
    sX[n * KPAD + k] = EnEvB[(size_t)k * PIX + p0 + n];
  }
  __syncthreads();

  int lane = threadIdx.x & 31;
  int mt   = threadIdx.x >> 5;
  int m = lane & 15, h = lane >> 4, n = m;
  v8f acc = {};
#pragma unroll
  for (int kk = 0; kk < 4; ++kk) {
    v16bf a = load_a_ptr(Wsp + (size_t)(mt * 16 + m) * CH + kk * 32, h);
    v16bf b = frag_b_lds(sX + n * KPAD + kk * 32 + h * 16);
    acc = wmma_bf16(a, b, acc);
  }
#pragma unroll
  for (int r = 0; r < 8; ++r) {
    size_t idx = (size_t)(mt * 16 + 8 * h + r) * PIX + p0 + n;
    float gsig = 1.0f / (1.0f + __expf(-acc[r]));
    float re   = EnRgb[idx];
    Fused[idx] = re + re * gsig + FeatRgb[idx];
  }
}

// ---------------------------------------------------------------------------
// final GEMM: out[64,P] = out_w[64,128] x normbf[128,P]
// 128 thr = 4 waves (M=64); blockIdx = pixel tile.
// ---------------------------------------------------------------------------
__global__ void __launch_bounds__(128)
k_gemm_out(const bf16_t* __restrict__ Wo, const bf16_t* __restrict__ Xbf,
           float* __restrict__ Out) {
  const int KPAD = 136;
  __shared__ __align__(16) bf16_t sX[16 * KPAD];
  int p0 = blockIdx.x * 16;
  for (int i = threadIdx.x; i < CH * 16; i += 128) {
    int k = i >> 4, n = i & 15;
    sX[n * KPAD + k] = Xbf[(size_t)k * PIX + p0 + n];
  }
  __syncthreads();

  int lane = threadIdx.x & 31;
  int mt   = threadIdx.x >> 5;   // 0..3
  int m = lane & 15, h = lane >> 4, n = m;
  v8f acc = {};
#pragma unroll
  for (int kk = 0; kk < 4; ++kk) {
    v16bf a = load_a_ptr(Wo + (size_t)(mt * 16 + m) * CH + kk * 32, h);
    v16bf b = frag_b_lds(sX + n * KPAD + kk * 32 + h * 16);
    acc = wmma_bf16(a, b, acc);
  }
#pragma unroll
  for (int r = 0; r < 8; ++r)
    Out[(size_t)(mt * 16 + 8 * h + r) * PIX + p0 + n] = acc[r];
}

// ---------------------------------------------------------------------------
// Host-side orchestration
// ---------------------------------------------------------------------------
extern "C" void kernel_launch(void* const* d_in, const int* in_sizes, int n_in,
                              void* d_out, int out_size, void* d_ws, size_t ws_size,
                              hipStream_t stream) {
  const float* rgb_high = (const float*)d_in[0];   // [64,P]
  const float* event_in = (const float*)d_in[1];   // [64,P]
  const float* conv_w   = (const float*)d_in[2];   // [128,64]
  const float* ln_g     = (const float*)d_in[3];   // [128]
  const float* ln_b     = (const float*)d_in[4];   // [128]
  const float* df1_w    = (const float*)d_in[5];   // [128,128,3,3]
  const float* df2_w    = (const float*)d_in[6];   // [3200,128]
  const float* sp_w     = (const float*)d_in[7];   // [128,128]
  const float* out_w    = (const float*)d_in[8];   // [64,128]
  float* out = (float*)d_out;                      // [64,P]

  char* ws = (char*)d_ws;
  size_t off = 0;
  auto alloc = [&](size_t bytes) -> char* {
    char* p = ws + off;
    off = (off + bytes + 255) & ~(size_t)255;
    return p;
  };

  bf16_t* wbf_conv = (bf16_t*)alloc((size_t)CH * CIN * 2);
  bf16_t* w9       = (bf16_t*)alloc((size_t)9 * CH * CH * 2);
  bf16_t* wbf_df2  = (bf16_t*)alloc((size_t)CH * 25 * CH * 2);
  bf16_t* wbf_sp   = (bf16_t*)alloc((size_t)CH * CH * 2);
  bf16_t* wbf_out  = (bf16_t*)alloc((size_t)CIN * CH * 2);
  bf16_t* xbf_rgb  = (bf16_t*)alloc((size_t)CIN * PIX * 2);
  bf16_t* xbf_ev   = (bf16_t*)alloc((size_t)CIN * PIX * 2);
  float*  Fr       = (float*) alloc((size_t)CH * PIX * 4);   // rgb feat (fp32)
  float*  Fe       = (float*) alloc((size_t)CH * PIX * 4);   // ev feat  (fp32)
  bf16_t* fbf_r    = (bf16_t*)alloc((size_t)CH * PIX * 2);   // rgb feat bf16
  bf16_t* fbf_e    = (bf16_t*)alloc((size_t)CH * PIX * 2);   // ev  feat bf16
  bf16_t* G_r      = (bf16_t*)alloc((size_t)CH * PIX * 2);   // relu(conv3x3)
  bf16_t* G_e      = (bf16_t*)alloc((size_t)CH * PIX * 2);
  float*  en_r     = (float*) alloc((size_t)CH * PIX * 4);   // rgb_en fp32
  bf16_t* en_e_bf  = (bf16_t*)alloc((size_t)CH * PIX * 2);   // ev_en bf16
  float*  fused    = (float*) alloc((size_t)CH * PIX * 4);
  bf16_t* fusedbf  = (bf16_t*)alloc((size_t)CH * PIX * 2);

  const int T = 256;
  auto grid1 = [](int n) { return (n + 255) / 256; };

  // 1. weight/input conversion
  k_f2bf<<<grid1(CH * CIN), T, 0, stream>>>(conv_w, wbf_conv, CH * CIN);
  k_repack_df1<<<grid1(CH * CH * 9), T, 0, stream>>>(df1_w, w9);
  k_f2bf<<<grid1(CH * 25 * CH), T, 0, stream>>>(df2_w, wbf_df2, CH * 25 * CH);
  k_f2bf<<<grid1(CH * CH), T, 0, stream>>>(sp_w, wbf_sp, CH * CH);
  k_f2bf<<<grid1(CIN * CH), T, 0, stream>>>(out_w, wbf_out, CIN * CH);
  k_f2bf<<<grid1(CIN * PIX), T, 0, stream>>>(rgb_high, xbf_rgb, CIN * PIX);
  k_f2bf<<<grid1(CIN * PIX), T, 0, stream>>>(event_in, xbf_ev, CIN * PIX);

  // 2. feat = conv1x1(x, conv_w)
  k_gemm_feat<<<PIX / 16, T, 0, stream>>>(wbf_conv, xbf_rgb, Fr);
  k_gemm_feat<<<PIX / 16, T, 0, stream>>>(wbf_conv, xbf_ev, Fe);

  // 3. LayerNorm (in-place fp32 + bf16 copy)
  k_ln<<<PIX / T, T, 0, stream>>>(Fr, ln_g, ln_b, fbf_r);
  k_ln<<<PIX / T, T, 0, stream>>>(Fe, ln_g, ln_b, fbf_e);

  // 4. G = relu(conv3x3(feat, df1_w))
  k_conv3x3<<<PIX / 16, T, 0, stream>>>(w9, fbf_r, G_r);
  k_conv3x3<<<PIX / 16, T, 0, stream>>>(w9, fbf_e, G_e);

  // 5. fused filter-generation GEMM + dynamic 5x5 filter application
  k_dynfilter<<<PIX / 16, T, 0, stream>>>(wbf_df2, G_r, Fr, en_r, (bf16_t*)nullptr);
  k_dynfilter<<<PIX / 16, T, 0, stream>>>(wbf_df2, G_e, Fe, (float*)nullptr, en_e_bf);

  // 6. fused = rgb_en * (1 + sigmoid(sp_w x ev_en)) + rgb_feat
  k_fuse<<<PIX / 16, T, 0, stream>>>(wbf_sp, en_e_bf, en_r, Fr, fused);

  // 7. LayerNorm on fused (bf16 copy feeds final GEMM)
  k_ln<<<PIX / T, T, 0, stream>>>(fused, ln_g, ln_b, fusedbf);

  // 8. out = conv1x1(norm, out_w)
  k_gemm_out<<<PIX / 16, 128, 0, stream>>>(wbf_out, fusedbf, out);
}
